// QLSTMGen_65481071396535
// MI455X (gfx1250) — matrix operand order repaired and checked by reading, not compiled, not run on hardware
//
#include <hip/hip_runtime.h>
#include <hip/hip_bf16.h>

// ---------------------------------------------------------------------------
// QLSTM for MI455X (gfx1250, wave32, WMMA 16x16x32 bf16, fp32 accum)
//   S=256 steps, B=64 batch, I=512 in, H=1024 hidden, NQ=4
// Round 3: async memory->LDS panel staging (GLOBAL_LOAD_ASYNC_TO_LDS_B128 +
// s_wait_asynccnt) on the recurrent critical path; near-cache prefetch.
// ---------------------------------------------------------------------------

#define S_LEN 256
#define BATCH 64
#define IN    512
#define HID   1024
#define GH    4096            // 4*HID
#define NQV   4
#define KT_G  48              // 1536/32 K-tiles in gate weights
#define KT_H  32              // 1024/32 K-tiles
#define TILE  512             // elements per packed 32x16 tile (32 lanes x 16)

typedef __attribute__((ext_vector_type(16))) __bf16 v16bf;
typedef __attribute__((ext_vector_type(8)))  float  v8f;

__device__ __forceinline__ v8f wmma_bf16(v16bf a, v16bf b, v8f c) {
    return __builtin_amdgcn_wmma_f32_16x16x32_bf16(false, a, false, b, (short)0, c, false, false);
}
__device__ __forceinline__ float sigf(float x) { return 1.0f / (1.0f + __expf(-x)); }

// Async 16B global->LDS copy (VGLOBAL: VDST = LDS byte addr, VADDR = 64b addr)
__device__ __forceinline__ void async_copy_b128(unsigned lds_off, const void* gptr) {
    asm volatile("global_load_async_to_lds_b128 %0, %1, off"
                 :: "v"(lds_off), "v"(gptr) : "memory");
}
__device__ __forceinline__ void wait_async0() {
    asm volatile("s_wait_asynccnt 0x0" ::: "memory");
}

// ---- fragment-order index helpers (CDNA5 wave32 layouts, CDNA5 ISA) ------
// B fragment: lane L, elem e  ->  (k_local, n_local) = (8*(L>>4)+(e&7)+16*(e>>3), L&15)
// A fragment: lane L, elem e  ->  (m_local, k_local) = (L&15, 8*(L>>4)+(e&7)+16*(e>>3))
__device__ __forceinline__ int bfrag_k(int lane, int e) { return ((lane >> 4) << 3) + (e & 7) + ((e >> 3) << 4); }
__device__ __forceinline__ int afrag_k(int lane, int e) {
    return ((lane >> 4) << 3) + (e & 7) + ((e >> 3) << 4);
}

// ---------------------------------------------------------------------------
// One-time prep: pack gate weights (1536x4096) into B-fragment order.
// Tile linear = nt*KT_G + kt  (nt-major => each wave's K-loop is contiguous)
// ---------------------------------------------------------------------------
__global__ void pack_wg_kernel(const float* __restrict__ Wf, const float* __restrict__ Wi,
                               const float* __restrict__ Wu, const float* __restrict__ Wo,
                               __bf16* __restrict__ Wt) {
    size_t idx = (size_t)blockIdx.x * blockDim.x + threadIdx.x;   // < 1536*4096
    int e    = (int)(idx & 15);
    int lane = (int)((idx >> 4) & 31);
    int tile = (int)(idx >> 9);
    int kt = tile % KT_G;
    int nt = tile / KT_G;
    int k   = kt * 32 + bfrag_k(lane, e);
    int col = nt * 16 + (lane & 15);
    int g = col >> 10, n = col & 1023;
    const float* src = (g == 0) ? Wf : (g == 1) ? Wi : (g == 2) ? Wu : Wo;
    Wt[idx] = (__bf16)src[(size_t)k * HID + n];
}

// Pack Wc (first 1024 rows) into B-fragment order; WcLast row fp32; BiasG; zero state
__global__ void prep_misc_kernel(const float* __restrict__ Wc,
                                 const float* __restrict__ bfv, const float* __restrict__ biv,
                                 const float* __restrict__ buv, const float* __restrict__ bov,
                                 __bf16* __restrict__ Wct, float* __restrict__ WcLast,
                                 float* __restrict__ BiasG, __bf16* __restrict__ hbf,
                                 float* __restrict__ cst) {
    int idx = blockIdx.x * blockDim.x + threadIdx.x;              // < 1024*1024
    {   // B-fragment pack of Wc[0:1024][0:1024]; tile linear = nt*KT_H + kt
        int e    = idx & 15;
        int lane = (idx >> 4) & 31;
        int tile = idx >> 9;
        int kt = tile & (KT_H - 1);
        int nt = tile >> 5;
        int k = kt * 32 + bfrag_k(lane, e);
        int n = nt * 16 + (lane & 15);
        Wct[idx] = (__bf16)Wc[(size_t)k * HID + n];
    }
    if (idx < HID) WcLast[idx] = Wc[(size_t)HID * HID + idx];
    if (idx < GH) {
        int g = idx >> 10, n = idx & 1023;
        const float* b = (g == 0) ? bfv : (g == 1) ? biv : (g == 2) ? buv : bov;
        BiasG[idx] = b[n];
    }
    if (idx < BATCH * HID) { hbf[idx] = (__bf16)0.0f; cst[idx] = 0.0f; }
}

// ---------------------------------------------------------------------------
// Parallel x-projection: Xproj[(s*B+b)][n] = x_row @ W[0:512] + BiasG[n]
// grid: 32768 blocks x 256 thr; block = (mt of 1024, group of 8 N-tiles)
// ---------------------------------------------------------------------------
__global__ void xproj_kernel(const float* __restrict__ x, const __bf16* __restrict__ Wt,
                             const float* __restrict__ BiasG, float* __restrict__ Xproj) {
    __shared__ __align__(32) __bf16 sA[16 * IN];      // A panel in fragment order
    int mt = blockIdx.x >> 5;          // 0..1023
    int ng = blockIdx.x & 31;          // 0..31
    const float* Ax = x + (size_t)mt * 16 * IN;       // 16 contiguous fp32 rows
    // gather-convert into A-fragment order: [kt][lane][e]
    for (int i = threadIdx.x; i < 16 * IN; i += blockDim.x) {
        int e = i & 15, lane = (i >> 4) & 31, kt = i >> 9;     // kt 0..15
        int m = lane & 15;
        int k = kt * 32 + afrag_k(lane, e);
        sA[i] = (__bf16)Ax[(size_t)m * IN + k];
    }
    __syncthreads();

    int w = threadIdx.x >> 5, lane = threadIdx.x & 31;
    int nt = ng * 8 + w;               // 0..255
    const __bf16* Bb = Wt + ((size_t)nt * KT_G) * TILE + (lane << 4);  // kt starts at 0
    const __bf16* Ab = sA + (lane << 4);
    v8f acc = {};
#pragma unroll 4
    for (int kt = 0; kt < IN / 32; ++kt) {
        __builtin_prefetch(Bb + (size_t)(kt + 4) * TILE, 0, 3);
        v16bf a = *(const v16bf*)(Ab + kt * TILE);
        v16bf b = *(const v16bf*)(Bb + (size_t)kt * TILE);
        acc = wmma_bf16(a, b, acc);
    }
    int n  = nt * 16 + (lane & 15);
    int mo = (lane >> 4) * 8;
    float bias = BiasG[n];
#pragma unroll
    for (int r = 0; r < 8; ++r)
        Xproj[(size_t)(mt * 16 + mo + r) * GH + n] = acc[r] + bias;
}

// ---------------------------------------------------------------------------
// Per-step gate pre-activation: pre = h @ W[512:1536] + Xproj[s]
// h is kept in A-fragment order: panel(mt) = 32 tiles x 512 = 32KB, linear.
// Panel staged via async global->LDS DMA (no VGPR round trip).
// grid: 128 blocks x 256 thr
// ---------------------------------------------------------------------------
__global__ void hgemm_pre_kernel(const __bf16* __restrict__ hbf, const __bf16* __restrict__ Wt,
                                 const float* __restrict__ Xproj, float* __restrict__ pre, int s) {
    __shared__ __align__(32) __bf16 sA[KT_H * TILE];  // 32KB packed A panel
    int mt = blockIdx.x >> 5;          // 0..3
    int ng = blockIdx.x & 31;          // 0..31
    {   // async linear 32KB copy: 256 threads x 8 x 16B
        const char* src = (const char*)(hbf + (size_t)mt * KT_H * TILE);
        unsigned dst = (unsigned)(size_t)(__bf16*)sA;
#pragma unroll
        for (int i = 0; i < 8; ++i) {
            unsigned off = (threadIdx.x + i * 256) * 16;
            async_copy_b128(dst + off, src + off);
        }
        wait_async0();
    }
    __syncthreads();

    int w = threadIdx.x >> 5, lane = threadIdx.x & 31;
    int nt = ng * 8 + w;               // 0..255
    const __bf16* Bb = Wt + ((size_t)nt * KT_G + 16) * TILE + (lane << 4);  // kt 16..47 = h rows
    const __bf16* Ab = sA + (lane << 4);
    v8f acc = {};
#pragma unroll 4
    for (int kt = 0; kt < KT_H; ++kt) {
        __builtin_prefetch(Bb + (size_t)(kt + 4) * TILE, 0, 3);
        v16bf a = *(const v16bf*)(Ab + kt * TILE);
        v16bf b = *(const v16bf*)(Bb + (size_t)kt * TILE);
        acc = wmma_bf16(a, b, acc);
    }
    int n  = nt * 16 + (lane & 15);
    int mo = (lane >> 4) * 8;
    const float* xp = Xproj + (size_t)s * BATCH * GH;
#pragma unroll
    for (int r = 0; r < 8; ++r) {
        int m = mt * 16 + mo + r;
        pre[(size_t)m * GH + n] = acc[r] + xp[(size_t)m * GH + n];
    }
}

// ---------------------------------------------------------------------------
// Quantum scalar per (gate,batch): q = cos(2*<pre[b, g*H : g*H+4], p_g>)
// ---------------------------------------------------------------------------
__global__ void qtheta_kernel(const float* __restrict__ pre,
                              const float* __restrict__ pf, const float* __restrict__ piq,
                              const float* __restrict__ pu, const float* __restrict__ po,
                              float* __restrict__ Qact) {
    int t = threadIdx.x;                         // g*64 + b
    if (t < 4 * BATCH) {
        int g = t >> 6, b = t & 63;
        const float* P = (g == 0) ? pf : (g == 1) ? piq : (g == 2) ? pu : po;
        const float* row = pre + (size_t)b * GH + g * HID;
        float th = 0.0f;
#pragma unroll
        for (int j = 0; j < NQV; ++j) th += row[j] * P[j];
        float q = __cosf(2.0f * th);
        Qact[t] = (g == 2) ? q : sigf(q);
    }
}

// Inner activations -> bf16 in A-fragment order. Panel (g,mt): 32 tiles x 512.
__global__ void acts_kernel(const float* __restrict__ pre, __bf16* __restrict__ Act) {
    int idx = blockIdx.x * blockDim.x + threadIdx.x;   // < 4*64*1024
    int e    = idx & 15;
    int lane = (idx >> 4) & 31;
    int tile = idx >> 9;               // 0..511
    int kt  = tile & (KT_H - 1);
    int mtg = tile >> 5;               // 0..15
    int g = mtg >> 2, mt = mtg & 3;
    int b = mt * 16 + (lane & 15);
    int k = kt * 32 + afrag_k(lane, e);
    float v = pre[(size_t)b * GH + g * HID + k];
    float a = (g == 2) ? tanhf(v) : sigf(v);
    Act[idx] = (__bf16)a;
}

// ---------------------------------------------------------------------------
// Comb GEMM: Comb[g][b][n] = act_g[b][:] @ Wc + bc[n] + Qact[g][b]*WcLast[n]
// grid: 128 blocks x 256 thr; block = (g 0..3, mt 0..3, group of 8 of 64 nt)
// ---------------------------------------------------------------------------
__global__ void comb_gemm_kernel(const __bf16* __restrict__ Act, const __bf16* __restrict__ Wct,
                                 const float* __restrict__ WcLast, const float* __restrict__ bc,
                                 const float* __restrict__ Qact, float* __restrict__ Comb) {
    __shared__ __align__(32) __bf16 sA[KT_H * TILE];
    int g   = blockIdx.x >> 5;         // 0..3
    int rem = blockIdx.x & 31;
    int mt  = rem >> 3;                // 0..3
    int ng  = rem & 7;                 // 0..7
    {   // async linear 32KB copy of packed Act panel (g,mt)
        const char* src = (const char*)(Act + ((size_t)(g * 4 + mt)) * KT_H * TILE);
        unsigned dst = (unsigned)(size_t)(__bf16*)sA;
#pragma unroll
        for (int i = 0; i < 8; ++i) {
            unsigned off = (threadIdx.x + i * 256) * 16;
            async_copy_b128(dst + off, src + off);
        }
        wait_async0();
    }
    __syncthreads();

    int w = threadIdx.x >> 5, lane = threadIdx.x & 31;
    int nt = ng * 8 + w;               // 0..63
    const __bf16* Bb = Wct + ((size_t)nt * KT_H) * TILE + (lane << 4);
    const __bf16* Ab = sA + (lane << 4);
    v8f acc = {};
#pragma unroll 4
    for (int kt = 0; kt < KT_H; ++kt) {
        __builtin_prefetch(Bb + (size_t)(kt + 4) * TILE, 0, 3);
        v16bf a = *(const v16bf*)(Ab + kt * TILE);
        v16bf b = *(const v16bf*)(Bb + (size_t)kt * TILE);
        acc = wmma_bf16(a, b, acc);
    }
    int n  = nt * 16 + (lane & 15);
    int mo = (lane >> 4) * 8;
    float wl = WcLast[n], bcn = bc[n];
#pragma unroll
    for (int r = 0; r < 8; ++r) {
        int b = mt * 16 + mo + r;
        Comb[((size_t)g * BATCH + b) * HID + n] = acc[r] + bcn + Qact[g * 64 + b] * wl;
    }
}

// ---------------------------------------------------------------------------
// Pointwise LSTM update; writes outputs[s]; rewrites h into A-fragment order.
// ---------------------------------------------------------------------------
__global__ void pointwise_kernel(const float* __restrict__ Comb, float* __restrict__ cst,
                                 __bf16* __restrict__ hbf, float* __restrict__ out,
                                 int s, int isLast) {
    int idx = blockIdx.x * blockDim.x + threadIdx.x;   // < 64*1024
    const int GB = BATCH * HID;                        // 65536
    int b = idx >> 10, k = idx & 1023;
    float fv = sigf(Comb[idx]);
    float iv = sigf(Comb[GB + idx]);
    float gv = tanhf(Comb[2 * GB + idx]);
    float ov = sigf(Comb[3 * GB + idx]);
    float c = fv * cst[idx] + iv * gv;
    float h = ov * tanhf(c);
    cst[idx] = c;
    // scatter h into A-fragment order: invert afrag mapping
    {
        int mt = b >> 4, m = b & 15;
        int kt = k >> 5, kl = k & 31;
        int half = (kl >> 4) & 1;      // which 16-block
        int r    = kl & 15;
        int hi   = r >> 3;             // lane half
        int e    = half * 8 + (r & 7);
        int lane = hi * 16 + m;
        hbf[((size_t)mt * KT_H + kt) * TILE + lane * 16 + e] = (__bf16)h;
    }
    out[(size_t)s * GB + idx] = h;
    if (isLast) {
        out[(size_t)S_LEN * GB + idx] = h;
        out[(size_t)S_LEN * GB + GB + idx] = c;
    }
}

// ---------------------------------------------------------------------------
extern "C" void kernel_launch(void* const* d_in, const int* in_sizes, int n_in,
                              void* d_out, int out_size, void* d_ws, size_t ws_size,
                              hipStream_t stream) {
    (void)in_sizes; (void)n_in; (void)out_size; (void)ws_size;
    const float* x   = (const float*)d_in[0];
    const float* Wf  = (const float*)d_in[1];
    const float* bfv = (const float*)d_in[2];
    const float* Wi  = (const float*)d_in[3];
    const float* biv = (const float*)d_in[4];
    const float* Wu  = (const float*)d_in[5];
    const float* buv = (const float*)d_in[6];
    const float* Wo  = (const float*)d_in[7];
    const float* bov = (const float*)d_in[8];
    const float* pf  = (const float*)d_in[9];
    const float* piq = (const float*)d_in[10];
    const float* pu  = (const float*)d_in[11];
    const float* po  = (const float*)d_in[12];
    const float* Wc  = (const float*)d_in[13];
    const float* bc  = (const float*)d_in[14];
    float* out = (float*)d_out;

    char* p = (char*)d_ws;
    auto carve = [&](size_t bytes) { char* r = p; p += (bytes + 255) & ~(size_t)255; return r; };
    __bf16* Wt     = (__bf16*)carve((size_t)(IN + HID) * GH * sizeof(__bf16));   // 12.6 MB packed
    __bf16* Wct    = (__bf16*)carve((size_t)HID * HID * sizeof(__bf16));         //  2.1 MB packed
    float*  WcLast = (float*) carve((size_t)HID * sizeof(float));
    float*  BiasG  = (float*) carve((size_t)GH * sizeof(float));
    __bf16* hbf    = (__bf16*)carve((size_t)BATCH * HID * sizeof(__bf16));       // packed A panels
    float*  cst    = (float*) carve((size_t)BATCH * HID * sizeof(float));
    float*  preb   = (float*) carve((size_t)BATCH * GH * sizeof(float));
    __bf16* Act    = (__bf16*)carve((size_t)4 * BATCH * HID * sizeof(__bf16));   // packed A panels
    float*  Qact   = (float*) carve((size_t)4 * BATCH * sizeof(float));
    float*  Comb   = (float*) carve((size_t)4 * BATCH * HID * sizeof(float));
    float*  Xproj  = (float*) carve((size_t)S_LEN * BATCH * GH * sizeof(float)); // 268 MB

    pack_wg_kernel<<<(1536 * 4096) / 256, 256, 0, stream>>>(Wf, Wi, Wu, Wo, Wt);
    prep_misc_kernel<<<(1024 * 1024) / 256, 256, 0, stream>>>(Wc, bfv, biv, buv, bov,
                                                              Wct, WcLast, BiasG, hbf, cst);
    xproj_kernel<<<32768, 256, 0, stream>>>(x, Wt, BiasG, Xproj);

    for (int s = 0; s < S_LEN; ++s) {
        hgemm_pre_kernel<<<128, 256, 0, stream>>>(hbf, Wt, Xproj, preb, s);
        qtheta_kernel<<<1, 256, 0, stream>>>(preb, pf, piq, pu, po, Qact);
        acts_kernel<<<1024, 256, 0, stream>>>(preb, Act);
        comb_gemm_kernel<<<128, 256, 0, stream>>>(Act, Wct, WcLast, bc, Qact, Comb);
        pointwise_kernel<<<256, 256, 0, stream>>>(Comb, cst, hbf, out, s, s == S_LEN - 1);
    }
}
